// GroupedQueryAttention_44495861187164
// MI455X (gfx1250) — matrix operand order repaired and checked by reading, not compiled, not run on hardware
//
#include <hip/hip_runtime.h>
#include <stdint.h>

// ---------------------------------------------------------------------------
// GQA forward for MI455X (gfx1250, wave32, WMMA bf16 -> f32 accumulate)
// T=4096, C=1024, H=16, HK=4, D=64, KV_DIM=256
// ---------------------------------------------------------------------------

typedef __bf16 bf16_t;
typedef __attribute__((ext_vector_type(16))) __bf16 v16bf;
typedef __attribute__((ext_vector_type(8)))  __bf16 v8bf;
typedef __attribute__((ext_vector_type(8)))  float  v8f;

#define SEQ_T 4096
#define DIM_C 1024
#define KV_DIM 256
#define N_HEADS 16
#define N_KVHEADS 4
#define HEAD_D 64

static __device__ __forceinline__ v16bf cat8(v8bf lo, v8bf hi) {
  return __builtin_shufflevector(lo, hi, 0,1,2,3,4,5,6,7,8,9,10,11,12,13,14,15);
}

static __device__ __forceinline__ v8f wmma_bf16(v16bf a, v16bf b, v8f c) {
  // D(f32 16x16) = A(bf16 16x32) * B(bf16 32x16) + C
  return __builtin_amdgcn_wmma_f32_16x16x32_bf16(
      /*neg_a=*/false, a, /*neg_b=*/false, b,
      /*c_mod=*/(short)0, c, /*reuse_a=*/false, /*reuse_b=*/false);
}

// A-fragment (16x32 bf16, row-per-lane): lanes 0-15 hold M=lane, K = k..k+15;
// lanes 16-31 hold M=lane-16, K = k+16..k+31.
static __device__ __forceinline__ v16bf load_a_frag(const bf16_t* rowptr, int k, int hi) {
  const bf16_t* p = rowptr + k + hi * 16;
  v8bf lo = *(const v8bf*)(p);
  v8bf hh = *(const v8bf*)(p + 8);
  return cat8(lo, hh);
}

// ---------------------------------------------------------------------------
// fp32 -> bf16 conversion
// ---------------------------------------------------------------------------
__global__ void cvt_f32_bf16(const float* __restrict__ src, bf16_t* __restrict__ dst, int n) {
  int i = blockIdx.x * blockDim.x + threadIdx.x;
  if (i < n) dst[i] = (bf16_t)src[i];
}

// ---------------------------------------------------------------------------
// Y[M,N] = A[M,K] * W[N,K]^T + bias[N]
// 8 waves/WG; each wave computes a 32x64 output tile (2 A-frags x 4 B-frags =
// 8 WMMA per K-chunk with full register reuse). Workgroup covers 64x256.
// Optionally writes bf16 copy (for downstream WMMA) and/or fp32 copy
// (for new_kv outputs with [Hkv,T,D] permutation, or the final output).
// ---------------------------------------------------------------------------
__global__ __launch_bounds__(256)
void gemm_xwT(const bf16_t* __restrict__ A, const bf16_t* __restrict__ W,
              const float* __restrict__ bias,
              bf16_t* __restrict__ out_bf, float* __restrict__ out_f32,
              int M, int N, int K, int kv_permute) {
  const int lane = threadIdx.x & 31;
  const int wave = threadIdx.x >> 5;
  const int hi   = (lane >> 4) & 1;
  const int l15  = lane & 15;

  const int m0 = blockIdx.y * 64  + (wave >> 2) * 32;   // 2 row-tiles of waves
  const int n0 = blockIdx.x * 256 + (wave & 3) * 64;    // 4 col-tiles of waves

  const bf16_t* arow0 = A + (size_t)(m0 + l15) * K;
  const bf16_t* arow1 = arow0 + (size_t)16 * K;
  const bf16_t* wr0   = W + (size_t)(n0 + l15) * K;
  const bf16_t* wr1   = wr0 + (size_t)16 * K;
  const bf16_t* wr2   = wr0 + (size_t)32 * K;
  const bf16_t* wr3   = wr0 + (size_t)48 * K;

  v8f c[2][4] = {};

  for (int k = 0; k < K; k += 32) {
    const int kb = k + hi * 16;
    v16bf a0 = load_a_frag(arow0, k, hi);
    v16bf a1 = load_a_frag(arow1, k, hi);
    // B-fragment (32x16, column-per-lane): lane half selects K 0-15 / 16-31.
    v16bf b0 = *(const v16bf*)(wr0 + kb);
    v16bf b1 = *(const v16bf*)(wr1 + kb);
    v16bf b2 = *(const v16bf*)(wr2 + kb);
    v16bf b3 = *(const v16bf*)(wr3 + kb);
    c[0][0] = wmma_bf16(a0, b0, c[0][0]);
    c[0][1] = wmma_bf16(a0, b1, c[0][1]);
    c[0][2] = wmma_bf16(a0, b2, c[0][2]);
    c[0][3] = wmma_bf16(a0, b3, c[0][3]);
    c[1][0] = wmma_bf16(a1, b0, c[1][0]);
    c[1][1] = wmma_bf16(a1, b1, c[1][1]);
    c[1][2] = wmma_bf16(a1, b2, c[1][2]);
    c[1][3] = wmma_bf16(a1, b3, c[1][3]);
  }

#pragma unroll
  for (int ni = 0; ni < 4; ++ni) {
    const int nc = n0 + ni * 16 + l15;
    const float bb = bias[nc];
#pragma unroll
    for (int mi = 0; mi < 2; ++mi) {
#pragma unroll
      for (int r = 0; r < 8; ++r) {
        const int m = m0 + mi * 16 + r + 8 * hi;  // C layout: halves -> M=r / M=r+8
        const float y = c[mi][ni][r] + bb;
        if (out_bf) out_bf[(size_t)m * N + nc] = (bf16_t)y;
        if (out_f32) {
          if (kv_permute) {  // [T, HK*D] -> [HK, T, D]
            out_f32[(((size_t)(nc >> 6) * M) + m) * HEAD_D + (nc & 63)] = y;
          } else {
            out_f32[(size_t)m * N + nc] = y;
          }
        }
      }
    }
  }
}

// ---------------------------------------------------------------------------
// Flash attention (no mask). 4 waves per WG, each wave = 16 q rows of one head.
// Grid: (T/64, N_HEADS). Q: bf16 [T, C] ([t][h*64+d]); K,V: bf16 [T, KV_DIM].
// Output O: bf16 [T, C]. K tile staged with gfx1250 async global->LDS copies.
// ---------------------------------------------------------------------------
#define KLDS_STRIDE 72   // 64 dims + pad (bf16 elems)
#define VLDS_STRIDE 40   // 32 keys + pad
#define PLDS_STRIDE 40   // 32 cols + pad

__global__ __launch_bounds__(128)
void attn_fwd(const bf16_t* __restrict__ Qb, const bf16_t* __restrict__ Kb,
              const bf16_t* __restrict__ Vb, bf16_t* __restrict__ Ob) {
  __shared__ __align__(32) bf16_t Klds[32 * KLDS_STRIDE];        // [key][dim]
  __shared__ __align__(32) bf16_t Vtlds[64 * VLDS_STRIDE];       // [dim][key] (transposed)
  __shared__ __align__(32) bf16_t Plds[4 * 16 * PLDS_STRIDE];    // per-wave P tile

  const int lane = threadIdx.x & 31;
  const int wave = threadIdx.x >> 5;
  const int hi   = (lane >> 4) & 1;
  const int l15  = lane & 15;

  const int h  = blockIdx.y;       // query head
  const int hk = h >> 2;           // kv head (N_GROUP = 4)
  const int q0 = blockIdx.x * 64 + wave * 16;

  // Q fragments for this wave's 16 rows, D=64 split into two K=32 chunks.
  const bf16_t* qrow = Qb + (size_t)(q0 + l15) * DIM_C + h * HEAD_D;
  const v16bf qf0 = load_a_frag(qrow, 0, hi);
  const v16bf qf1 = load_a_frag(qrow, 32, hi);

  v8f acc0 = {}, acc1 = {}, acc2 = {}, acc3 = {};
  float m_run[8], l_run[8];
#pragma unroll
  for (int r = 0; r < 8; ++r) { m_run[r] = -3.0e38f; l_run[r] = 0.0f; }

  bf16_t* Pw = Plds + wave * 16 * PLDS_STRIDE;

  const int key = threadIdx.x >> 2;   // 0..31 (cooperative loader index)
  const int seg = threadIdx.x & 3;    // 16-dim segment

  for (int j0 = 0; j0 < SEQ_T; j0 += 32) {
    // ---- stage K tile via async global->LDS (ASYNCcnt), V tile transposed ----
    {
      const bf16_t* ksrc = Kb + (size_t)(j0 + key) * KV_DIM + hk * HEAD_D + seg * 16;
      uint32_t koff = (uint32_t)(uintptr_t)(Klds + key * KLDS_STRIDE + seg * 16);
      uint64_t kga  = (uint64_t)(uintptr_t)ksrc;
      asm volatile("global_load_async_to_lds_b128 %0, %1, off"
                   :: "v"(koff), "v"(kga) : "memory");

      const bf16_t* vsrc = Vb + (size_t)(j0 + key) * KV_DIM + hk * HEAD_D + seg * 16;
      v16bf vv = *(const v16bf*)vsrc;
#pragma unroll
      for (int i = 0; i < 16; ++i)
        Vtlds[(seg * 16 + i) * VLDS_STRIDE + key] = vv[i];

      asm volatile("s_wait_asynccnt 0x0" ::: "memory");
    }
    __syncthreads();

    if (j0 + 32 < SEQ_T) {
      __builtin_prefetch(Kb + (size_t)(j0 + 32 + key) * KV_DIM + hk * HEAD_D, 0, 1);
      __builtin_prefetch(Vb + (size_t)(j0 + 32 + key) * KV_DIM + hk * HEAD_D, 0, 1);
    }

    // ---- S = Q * K^T  (16 x 32, two 16x16 N-tiles) ----
    v8f s0 = {}, s1 = {};
    {
      v16bf b;
      b = *(const v16bf*)(Klds + (0  + l15) * KLDS_STRIDE + 0  + hi * 16);
      s0 = wmma_bf16(qf0, b, s0);
      b = *(const v16bf*)(Klds + (0  + l15) * KLDS_STRIDE + 32 + hi * 16);
      s0 = wmma_bf16(qf1, b, s0);
      b = *(const v16bf*)(Klds + (16 + l15) * KLDS_STRIDE + 0  + hi * 16);
      s1 = wmma_bf16(qf0, b, s1);
      b = *(const v16bf*)(Klds + (16 + l15) * KLDS_STRIDE + 32 + hi * 16);
      s1 = wmma_bf16(qf1, b, s1);
    }

    // ---- online softmax update (rows live per-VGPR, N across 16-lane halves) ----
    float rmax[8], rsum[8], p0[8], p1[8], factor[8];
#pragma unroll
    for (int r = 0; r < 8; ++r) {
      s0[r] *= 0.125f;   // 1/sqrt(64)
      s1[r] *= 0.125f;
      rmax[r] = fmaxf(s0[r], s1[r]);
    }
#pragma unroll
    for (int off = 1; off < 16; off <<= 1) {
#pragma unroll
      for (int r = 0; r < 8; ++r)
        rmax[r] = fmaxf(rmax[r], __shfl_xor(rmax[r], off, 32));
    }
#pragma unroll
    for (int r = 0; r < 8; ++r) {
      const float mn = fmaxf(m_run[r], rmax[r]);
      factor[r] = __expf(m_run[r] - mn);
      p0[r] = __expf(s0[r] - mn);
      p1[r] = __expf(s1[r] - mn);
      rsum[r] = p0[r] + p1[r];
      m_run[r] = mn;
    }
#pragma unroll
    for (int off = 1; off < 16; off <<= 1) {
#pragma unroll
      for (int r = 0; r < 8; ++r)
        rsum[r] += __shfl_xor(rsum[r], off, 32);
    }
#pragma unroll
    for (int r = 0; r < 8; ++r) {
      l_run[r] = l_run[r] * factor[r] + rsum[r];
      acc0[r] *= factor[r];
      acc1[r] *= factor[r];
      acc2[r] *= factor[r];
      acc3[r] *= factor[r];
    }

    // ---- C-layout -> A-layout for P via LDS ----
#pragma unroll
    for (int r = 0; r < 8; ++r) {
      const int row = r + 8 * hi;
      Pw[row * PLDS_STRIDE + l15]      = (bf16_t)p0[r];
      Pw[row * PLDS_STRIDE + 16 + l15] = (bf16_t)p1[r];
    }
    v16bf pf;
    {
      v8bf lo = *(const v8bf*)(Pw + l15 * PLDS_STRIDE + hi * 16);
      v8bf hh = *(const v8bf*)(Pw + l15 * PLDS_STRIDE + hi * 16 + 8);
      pf = cat8(lo, hh);
    }

    // ---- acc += P * V  (16 x 64, four 16x16 N-tiles over head dim) ----
    {
      v16bf bv;
      bv = *(const v16bf*)(Vtlds + (0  + l15) * VLDS_STRIDE + hi * 16);
      acc0 = wmma_bf16(pf, bv, acc0);
      bv = *(const v16bf*)(Vtlds + (16 + l15) * VLDS_STRIDE + hi * 16);
      acc1 = wmma_bf16(pf, bv, acc1);
      bv = *(const v16bf*)(Vtlds + (32 + l15) * VLDS_STRIDE + hi * 16);
      acc2 = wmma_bf16(pf, bv, acc2);
      bv = *(const v16bf*)(Vtlds + (48 + l15) * VLDS_STRIDE + hi * 16);
      acc3 = wmma_bf16(pf, bv, acc3);
    }
    __syncthreads();
  }

  // ---- normalize and write O (bf16, [t][h*64+d]) ----
#pragma unroll
  for (int r = 0; r < 8; ++r) {
    const int m = q0 + r + 8 * hi;
    const float inv = 1.0f / l_run[r];
    bf16_t* orow = Ob + (size_t)m * DIM_C + h * HEAD_D;
    orow[0  + l15] = (bf16_t)(acc0[r] * inv);
    orow[16 + l15] = (bf16_t)(acc1[r] * inv);
    orow[32 + l15] = (bf16_t)(acc2[r] * inv);
    orow[48 + l15] = (bf16_t)(acc3[r] * inv);
  }
}

// ---------------------------------------------------------------------------
// Host-side orchestration
// ---------------------------------------------------------------------------
extern "C" void kernel_launch(void* const* d_in, const int* in_sizes, int n_in,
                              void* d_out, int out_size, void* d_ws, size_t ws_size,
                              hipStream_t stream) {
  const float* x  = (const float*)d_in[0];
  const float* Wq = (const float*)d_in[1];
  const float* bq = (const float*)d_in[2];
  const float* Wk = (const float*)d_in[3];
  const float* bk = (const float*)d_in[4];
  const float* Wv = (const float*)d_in[5];
  const float* bv = (const float*)d_in[6];
  const float* Wo = (const float*)d_in[7];
  const float* bo = (const float*)d_in[8];

  float* out  = (float*)d_out;                              // [T, C]
  float* outK = out + (size_t)SEQ_T * DIM_C;                // [HK, T, D]
  float* outV = outK + (size_t)N_KVHEADS * SEQ_T * HEAD_D;  // [HK, T, D]

  // workspace layout (bf16 staging buffers, 256B-aligned offsets)
  char* ws = (char*)d_ws;
  bf16_t* x_bf  = (bf16_t*)(ws + 0);                        //  8 MiB
  bf16_t* q_bf  = (bf16_t*)(ws + 8388608);                  //  8 MiB
  bf16_t* o_bf  = (bf16_t*)(ws + 16777216);                 //  8 MiB
  bf16_t* k_bf  = (bf16_t*)(ws + 25165824);                 //  2 MiB
  bf16_t* v_bf  = (bf16_t*)(ws + 27262976);                 //  2 MiB
  bf16_t* wq_bf = (bf16_t*)(ws + 29360128);                 //  2 MiB
  bf16_t* wo_bf = (bf16_t*)(ws + 31457280);                 //  2 MiB
  bf16_t* wk_bf = (bf16_t*)(ws + 33554432);                 //  512 KiB
  bf16_t* wv_bf = (bf16_t*)(ws + 34078720);                 //  512 KiB

  auto cvt = [&](const float* s, bf16_t* d, int n) {
    cvt_f32_bf16<<<(n + 255) / 256, 256, 0, stream>>>(s, d, n);
  };

  cvt(x,  x_bf,  SEQ_T * DIM_C);
  cvt(Wq, wq_bf, DIM_C * DIM_C);
  cvt(Wk, wk_bf, KV_DIM * DIM_C);
  cvt(Wv, wv_bf, KV_DIM * DIM_C);
  cvt(Wo, wo_bf, DIM_C * DIM_C);

  // Projections: Q (bf16 only), K/V (bf16 + permuted fp32 new_kv outputs)
  gemm_xwT<<<dim3(DIM_C / 256, SEQ_T / 64), 256, 0, stream>>>(
      x_bf, wq_bf, bq, q_bf, nullptr, SEQ_T, DIM_C, DIM_C, 0);
  gemm_xwT<<<dim3(KV_DIM / 256, SEQ_T / 64), 256, 0, stream>>>(
      x_bf, wk_bf, bk, k_bf, outK, SEQ_T, KV_DIM, DIM_C, 1);
  gemm_xwT<<<dim3(KV_DIM / 256, SEQ_T / 64), 256, 0, stream>>>(
      x_bf, wv_bf, bv, v_bf, outV, SEQ_T, KV_DIM, DIM_C, 1);

  // Flash attention over all heads
  attn_fwd<<<dim3(SEQ_T / 64, N_HEADS), 128, 0, stream>>>(q_bf, k_bf, v_bf, o_bf);

  // Output projection -> fp32 d_out
  gemm_xwT<<<dim3(DIM_C / 256, SEQ_T / 64), 256, 0, stream>>>(
      o_bf, wo_bf, bo, nullptr, out, SEQ_T, DIM_C, DIM_C, 0);
}
